// DyGraphAtt2d_40252433498610
// MI455X (gfx1250) — matrix operand order repaired and checked by reading, not compiled, not run on hardware
//
#include <hip/hip_runtime.h>
#include <math.h>

// ---------------------------------------------------------------------------
// DyGraphAtt2d: per-batch kNN graph (k=16) in feature space + GATConv
//   B=8, C=128, N=4096, H=4, F=128, HF=512, K=16
// Strategy: f16-input / f32-accum WMMA (v_wmma_f32_16x16x32_f16) for both the
// Gram matrix (fused with top-k, never materialized to HBM) and h = xf @ W.
// Top-k selection is parallelized 8-way per row (128 threads total) with a
// final per-row merge, so the knn kernel is WMMA-bound, not scan-bound.
// ---------------------------------------------------------------------------

#define NB 8
#define NC 128
#define NN 4096
#define NH 4
#define NF 128
#define NHF 512
#define NK 16

typedef __attribute__((ext_vector_type(16))) _Float16 v16h;
typedef __attribute__((ext_vector_type(8)))  float    v8f;

__device__ __forceinline__ v8f wmma_f16(v16h a, v16h b, v8f c) {
  // D = A(16x32 f16) * B(32x16 f16) + C(16x16 f32)
  return __builtin_amdgcn_wmma_f32_16x16x32_f16(
      /*neg_a=*/false, a, /*neg_b=*/false, b,
      /*c_mod=*/(short)0, c, /*reuse_a=*/false, /*reuse_b=*/false);
}

// A-matrix 16x32 f16 fragment, row-major source [rows, ld].
// ISA layout: lanes 0-15 row m=lane: v0..v3 = K 0..7 pairs, v4..v7 = K 16..23;
// lanes 16-31 same rows, K+8 in each group.
__device__ __forceinline__ v16h load_frag_a(const _Float16* p, int ld,
                                            int row0, int kbase, int lane) {
  const int m = lane & 15, kh = lane >> 4;
  const _Float16* base = p + (size_t)(row0 + m) * ld + kbase;
  v16h out;
#pragma unroll
  for (int i = 0; i < 8; ++i) {
    const int k0 = ((i < 4) ? 0 : 16) + kh * 8 + (i & 3) * 2;
    out[2 * i]     = base[k0];
    out[2 * i + 1] = base[k0 + 1];
  }
  return out;
}

// B-matrix 32x16 f16 fragment. Source is "column-major over K contiguous":
// p[(col)*ld + k]. Lane n = col, lanes 0-15: v0..v3 = K 0..7, v4..v7 = K 16..23;
// lanes 16-31 K+8 in each group (same as A).
__device__ __forceinline__ v16h load_frag_b(const _Float16* p, int ld,
                                            int col0, int kbase, int lane) {
  const int n = lane & 15, kh = lane >> 4;
  const _Float16* base = p + (size_t)(col0 + n) * ld + kbase;
  v16h out;
#pragma unroll
  for (int v = 0; v < 8; ++v) {
    const int k0 = ((v < 4) ? 0 : 16) + kh * 8 + (v & 3) * 2;
    out[2 * v]     = base[k0];
    out[2 * v + 1] = base[k0 + 1];
  }
  return out;
}

// ---------------------------------------------------------------------------
// 1) x [B,C,N] -> xf16 [B*N, C] (f16) + sq norms [B*N]
// ---------------------------------------------------------------------------
__global__ void prep_x_kernel(const float* __restrict__ x,
                              _Float16* __restrict__ xf16,
                              float* __restrict__ sq) {
  const int idx = blockIdx.x * blockDim.x + threadIdx.x;   // global node
  const int b = idx >> 12, n = idx & (NN - 1);
  const float* xb = x + (size_t)b * NC * NN;
  float s = 0.f;
#pragma unroll 4
  for (int c = 0; c < NC; ++c) {
    const float v = xb[(size_t)c * NN + n];                // coalesced over n
    s += v * v;
    xf16[(size_t)idx * NC + c] = (_Float16)v;
  }
  sq[idx] = s;
}

// 2) W [C, HF] -> Wt16 [HF, C] (f16), so B-frag pairs are contiguous.
__global__ void prep_w_kernel(const float* __restrict__ W,
                              _Float16* __restrict__ wt16) {
  const int hf = blockIdx.x * blockDim.x + threadIdx.x;
  if (hf >= NHF) return;
#pragma unroll 4
  for (int c = 0; c < NC; ++c)
    wt16[(size_t)hf * NC + c] = (_Float16)W[(size_t)c * NHF + hf];
}

// ---------------------------------------------------------------------------
// 3) h = xf @ W : [32768 x 512], K=128. Block = 4 waves, each wave a 16x16 tile.
// ---------------------------------------------------------------------------
__global__ void h_gemm_kernel(const _Float16* __restrict__ xf16,
                              const _Float16* __restrict__ wt16,
                              float* __restrict__ h) {
  const int row0 = blockIdx.x * 16;
  const int wave = threadIdx.x >> 5, lane = threadIdx.x & 31;
  const int col0 = blockIdx.y * 64 + wave * 16;
  v8f acc = {};
#pragma unroll
  for (int kb = 0; kb < NC; kb += 32)
    acc = wmma_f16(load_frag_a(xf16, NC, row0, kb, lane),
                   load_frag_b(wt16, NC, col0, kb, lane), acc);
  // C/D layout: lane L holds col n=L&15; VGPR r holds row m = 8*(L>>4)+r
  const int n  = col0 + (lane & 15);
  const int mh = (lane >> 4) * 8;
#pragma unroll
  for (int r = 0; r < 8; ++r)
    h[(size_t)(row0 + mh + r) * NHF + n] = acc[r];
}

// ---------------------------------------------------------------------------
// 4) a_src/a_dst: wave per node, shuffle-reduced dots over F=128 per head.
// ---------------------------------------------------------------------------
__global__ void attn_kernel(const float* __restrict__ h,
                            const float* __restrict__ att_src,
                            const float* __restrict__ att_dst,
                            float* __restrict__ asrc,
                            float* __restrict__ adst) {
  const int wave = threadIdx.x >> 5, lane = threadIdx.x & 31;
  const int bn = blockIdx.x * 8 + wave;
  const float* hp = h + (size_t)bn * NHF;
#pragma unroll
  for (int hh = 0; hh < NH; ++hh) {
    float s = 0.f, d = 0.f;
#pragma unroll
    for (int i = 0; i < 4; ++i) {
      const int f = lane + 32 * i;
      const float v = hp[hh * NF + f];
      s += v * att_src[hh * NF + f];
      d += v * att_dst[hh * NF + f];
    }
#pragma unroll
    for (int off = 16; off > 0; off >>= 1) {
      s += __shfl_xor(s, off, 32);
      d += __shfl_xor(d, off, 32);
    }
    if (lane == 0) { asrc[bn * NH + hh] = s; adst[bn * NH + hh] = d; }
  }
}

// ---------------------------------------------------------------------------
// 5) Fused Gram + top-16. Block = 4 waves handles 16 query rows of one batch.
//    Per 64-column chunk: each wave computes a 16x16 dot tile (4 WMMA, K=128)
//    and writes sq[col]-2*dot into LDS (||q||^2 dropped: constant per row,
//    does not affect ranking). Then ALL 128 threads select: thread (row, seg)
//    scans its 8-column stripe into a private sorted top-16; a final per-row
//    merge of the 8 partial lists produces the neighbor list.
// ---------------------------------------------------------------------------
__global__ void knn_kernel(const _Float16* __restrict__ xf16,
                           const float* __restrict__ sq,
                           int* __restrict__ nbr) {
  const int b    = blockIdx.x >> 8;        // 256 row-tiles per batch
  const int row0 = (blockIdx.x & 255) * 16;
  const int tid  = threadIdx.x;
  const int wave = tid >> 5, lane = tid & 31;
  const int row  = tid & 15, seg = tid >> 4;   // selection role: 8 segs/row
  __shared__ float distL[16][65];              // +1 pad: conflict-free scans
  __shared__ float pd[16][8][NK];              // per-row partial top-16 dists
  __shared__ int   pi[16][8][NK];              // per-row partial top-16 idxs

  const _Float16* Xb  = xf16 + (size_t)b * NN * NC;
  const float*    sqb = sq + (size_t)b * NN;

  // Query fragments are loop-invariant: hoist all 4 K-chunks.
  const v16h a0 = load_frag_a(Xb, NC, row0,  0, lane);
  const v16h a1 = load_frag_a(Xb, NC, row0, 32, lane);
  const v16h a2 = load_frag_a(Xb, NC, row0, 64, lane);
  const v16h a3 = load_frag_a(Xb, NC, row0, 96, lane);

  float kd[NK];
  int   ki[NK];
#pragma unroll
  for (int p = 0; p < NK; ++p) { kd[p] = 3.0e38f; ki[p] = 0; }

  for (int base = 0; base < NN; base += 64) {
    const int j0 = base + wave * 16;
    v8f acc = {};
    acc = wmma_f16(a0, load_frag_b(Xb, NC, j0,  0, lane), acc);
    acc = wmma_f16(a1, load_frag_b(Xb, NC, j0, 32, lane), acc);
    acc = wmma_f16(a2, load_frag_b(Xb, NC, j0, 64, lane), acc);
    acc = wmma_f16(a3, load_frag_b(Xb, NC, j0, 96, lane), acc);

    const float sqc = sqb[j0 + (lane & 15)];
    const int   mh  = (lane >> 4) * 8;
#pragma unroll
    for (int r = 0; r < 8; ++r)
      distL[mh + r][wave * 16 + (lane & 15)] = sqc - 2.0f * acc[r];
    __syncthreads();

    // Parallel selection: each thread scans its 8-column stripe of its row.
#pragma unroll
    for (int c = 0; c < 8; ++c) {
      float v  = distL[row][seg * 8 + c];
      int   vi = base + seg * 8 + c;
      if (v < kd[NK - 1]) {
#pragma unroll
        for (int p = 0; p < NK; ++p) {
          if (v < kd[p]) {
            const float td = kd[p]; const int ti = ki[p];
            kd[p] = v; ki[p] = vi; v = td; vi = ti;
          }
        }
      }
    }
    __syncthreads();
  }

  // Publish partial lists, then one thread per row merges 8x16 -> top-16.
#pragma unroll
  for (int p = 0; p < NK; ++p) { pd[row][seg][p] = kd[p]; pi[row][seg][p] = ki[p]; }
  __syncthreads();

  if (tid < 16) {
    float fd[NK]; int fi[NK];
#pragma unroll
    for (int p = 0; p < NK; ++p) { fd[p] = pd[tid][0][p]; fi[p] = pi[tid][0][p]; }
    for (int s = 1; s < 8; ++s) {
#pragma unroll
      for (int p = 0; p < NK; ++p) {
        float v  = pd[tid][s][p];
        int   vi = pi[tid][s][p];
        if (v < fd[NK - 1]) {
#pragma unroll
          for (int q = 0; q < NK; ++q) {
            if (v < fd[q]) {
              const float td = fd[q]; const int ti = fi[q];
              fd[q] = v; fi[q] = vi; v = td; vi = ti;
            }
          }
        }
      }
    }
    int* dst = nbr + (size_t)(b * NN + row0 + tid) * NK;
#pragma unroll
    for (int p = 0; p < NK; ++p) dst[p] = fi[p];
  }
}

// ---------------------------------------------------------------------------
// 6) Per-node GAT aggregation: leaky_relu -> softmax over k -> weighted sum of
//    neighbor features, mean over heads, + bias. Block(128) = one node; the
//    f-dimension gathers of h are fully coalesced; h[b] (8.4 MB) lives in L2.
// ---------------------------------------------------------------------------
__global__ void aggregate_kernel(const float* __restrict__ h,
                                 const float* __restrict__ asrc,
                                 const float* __restrict__ adst,
                                 const int* __restrict__ nbr,
                                 const float* __restrict__ bias,
                                 float* __restrict__ tmp) {
  const int bn = blockIdx.x;
  const int b  = bn >> 12;
  const int tid = threadIdx.x;
  __shared__ int   nb[NK];
  __shared__ float eL[NK][NH];
  __shared__ float alpha[NK][NH];

  if (tid < NK) nb[tid] = nbr[(size_t)bn * NK + tid];
  __syncthreads();

  if (tid < NK * NH) {
    const int kk = tid >> 2, hh = tid & 3;
    const float v = asrc[(size_t)(b * NN + nb[kk]) * NH + hh] +
                    adst[(size_t)bn * NH + hh];
    eL[kk][hh] = (v > 0.f) ? v : 0.2f * v;     // leaky_relu(0.2)
  }
  __syncthreads();

  if (tid < NH) {
    float m = -3.0e38f;
    for (int kk = 0; kk < NK; ++kk) m = fmaxf(m, eL[kk][tid]);
    float s = 0.f;
    for (int kk = 0; kk < NK; ++kk) {
      const float t = __expf(eL[kk][tid] - m);
      alpha[kk][tid] = t; s += t;
    }
    const float inv = 1.f / s;
    for (int kk = 0; kk < NK; ++kk) alpha[kk][tid] *= inv;
  }
  __syncthreads();

  float acc = 0.f;
#pragma unroll
  for (int hh = 0; hh < NH; ++hh) {
#pragma unroll 4
    for (int kk = 0; kk < NK; ++kk) {
      acc += alpha[kk][hh] *
             h[(size_t)(b * NN + nb[kk]) * NHF + hh * NF + tid];
    }
  }
  tmp[(size_t)bn * NF + tid] = acc * 0.25f + bias[tid];   // mean over 4 heads
}

// ---------------------------------------------------------------------------
// 7) [B,N,F] -> [B,F,N] tiled transpose so final HBM stores are coalesced.
// ---------------------------------------------------------------------------
__global__ void transpose_out_kernel(const float* __restrict__ tmp,
                                     float* __restrict__ out) {
  __shared__ float t[32][33];
  const int b  = blockIdx.z;
  const int n0 = blockIdx.x * 32, f0 = blockIdx.y * 32;
  const int ix = threadIdx.x, ty = threadIdx.y;            // 32 x 8
  const float* src = tmp + ((size_t)b * NN + n0) * NF + f0;
#pragma unroll
  for (int i = 0; i < 32; i += 8)
    t[ty + i][ix] = src[(size_t)(ty + i) * NF + ix];       // t[n][f]
  __syncthreads();
  float* dst = out + ((size_t)b * NF + f0) * NN + n0;
#pragma unroll
  for (int i = 0; i < 32; i += 8)
    dst[(size_t)(ty + i) * NN + ix] = t[ix][ty + i];
}

// ---------------------------------------------------------------------------
extern "C" void kernel_launch(void* const* d_in, const int* in_sizes, int n_in,
                              void* d_out, int out_size, void* d_ws, size_t ws_size,
                              hipStream_t stream) {
  const float* x       = (const float*)d_in[0];   // [B,C,N,1]
  const float* W       = (const float*)d_in[1];   // [C,H*F]
  const float* att_src = (const float*)d_in[2];   // [H,F]
  const float* att_dst = (const float*)d_in[3];   // [H,F]
  const float* bias    = (const float*)d_in[4];   // [F]
  (void)in_sizes; (void)n_in; (void)out_size; (void)ws_size;

  char* ws = (char*)d_ws;
  _Float16* xf16 = (_Float16*)ws;  ws += (size_t)NB * NN * NC * sizeof(_Float16);
  _Float16* wt16 = (_Float16*)ws;  ws += (size_t)NHF * NC * sizeof(_Float16);
  float*    sq   = (float*)ws;     ws += (size_t)NB * NN * sizeof(float);
  float*    hbuf = (float*)ws;     ws += (size_t)NB * NN * NHF * sizeof(float);
  float*    asrc = (float*)ws;     ws += (size_t)NB * NN * NH * sizeof(float);
  float*    adst = (float*)ws;     ws += (size_t)NB * NN * NH * sizeof(float);
  int*      nbr  = (int*)ws;       ws += (size_t)NB * NN * NK * sizeof(int);
  float*    tmp  = (float*)ws;     // [B,N,F]

  prep_x_kernel<<<(NB * NN) / 256, 256, 0, stream>>>(x, xf16, sq);
  prep_w_kernel<<<2, 256, 0, stream>>>(W, wt16);
  h_gemm_kernel<<<dim3((NB * NN) / 16, NHF / 64), 128, 0, stream>>>(xf16, wt16, hbuf);
  attn_kernel<<<(NB * NN) / 8, 256, 0, stream>>>(hbuf, att_src, att_dst, asrc, adst);
  knn_kernel<<<NB * (NN / 16), 128, 0, stream>>>(xf16, sq, nbr);
  aggregate_kernel<<<NB * NN, 128, 0, stream>>>(hbuf, asrc, adst, nbr, bias, tmp);
  transpose_out_kernel<<<dim3(NN / 32, NF / 32, NB), dim3(32, 8), 0, stream>>>(
      tmp, (float*)d_out);
}
